// GCNMeanMix_49323404427794
// MI455X (gfx1250) — compile-verified, matrix operand order checked by reading, not run on hardware
//
#include <hip/hip_runtime.h>
#include <hip/hip_bf16.h>

typedef _Float16 v16h __attribute__((ext_vector_type(16)));
typedef _Float16 v8h  __attribute__((ext_vector_type(8)));
typedef _Float16 v4h  __attribute__((ext_vector_type(4)));
typedef float    v8f  __attribute__((ext_vector_type(8)));

#define H 128
#define NN 128
#define LDST 136           // LDS row stride in halves: 272B = 17*16B, keeps b128 alignment, avoids bank conflicts
#define LDS_MAT (128 * LDST)
#define LDS_BYTES (4u * LDS_MAT * 2u)

union FragH { v16h v; v8h h[2]; };

// Load a 16x32 WMMA operand fragment.
// Works for A (row-major [m][k]) and for B when stored N-major ([n][k]).
// ISA layout (16-bit A, wave32): lane L -> row = L%16;
//   VGPR0..3 hold K {0..7} (lanes 0-15) / {8..15} (lanes 16-31); VGPR4..7 add +16.
__device__ __forceinline__ v16h load_frag(const _Float16* __restrict__ base,
                                          int tile, int kt, int lane, int stride) {
  const int r  = tile * 16 + (lane & 15);
  const int kb = kt * 32 + ((lane >> 4) & 1) * 8;
  const _Float16* p = base + r * stride + kb;
  FragH f;
  f.h[0] = *(const v8h*)(p);
  f.h[1] = *(const v8h*)(p + 16);
  return f.v;
}

// D(col-major f16, raw) = A(128x128) * B(128x128); A row-major LDS, B N-major.
// Column-major store => next GEMM can consume D directly as a B operand.
__device__ __forceinline__ void gemm_colmajor(const _Float16* __restrict__ A, int strideA,
                                              const _Float16* __restrict__ B, int strideB,
                                              _Float16* __restrict__ Dt, int wave, int lane) {
#pragma unroll
  for (int nt = 0; nt < 8; ++nt) {
    v8f acc = {0.f, 0.f, 0.f, 0.f, 0.f, 0.f, 0.f, 0.f};
#pragma unroll
    for (int kt = 0; kt < 4; ++kt) {
      v16h a = load_frag(A, wave, kt, lane, strideA);
      v16h b = load_frag(B, nt,  kt, lane, strideB);
      acc = __builtin_amdgcn_wmma_f32_16x16x32_f16(false, a, false, b, (short)0, acc, false, false);
    }
    const int n  = nt * 16 + (lane & 15);
    const int m0 = wave * 16 + ((lane >> 4) & 1) * 8;
    v8h out;
#pragma unroll
    for (int r = 0; r < 8; ++r) out[r] = (_Float16)acc[r];
    *(v8h*)(Dt + n * LDST + m0) = out;   // contiguous 16B ds_store
  }
}

// D(row-major f16) = relu(A*B + bias); used for Anorm @ (xW) + b
__device__ __forceinline__ void gemm_bias_relu_row(const _Float16* __restrict__ A, int strideA,
                                                   const _Float16* __restrict__ B, int strideB,
                                                   const float* __restrict__ bias,
                                                   _Float16* __restrict__ D, int wave, int lane) {
#pragma unroll
  for (int nt = 0; nt < 8; ++nt) {
    v8f acc = {0.f, 0.f, 0.f, 0.f, 0.f, 0.f, 0.f, 0.f};
#pragma unroll
    for (int kt = 0; kt < 4; ++kt) {
      v16h a = load_frag(A, wave, kt, lane, strideA);
      v16h b = load_frag(B, nt,  kt, lane, strideB);
      acc = __builtin_amdgcn_wmma_f32_16x16x32_f16(false, a, false, b, (short)0, acc, false, false);
    }
    const int n  = nt * 16 + (lane & 15);
    const int m0 = wave * 16 + ((lane >> 4) & 1) * 8;
    const float bn = bias[n];
#pragma unroll
    for (int r = 0; r < 8; ++r)
      D[(m0 + r) * LDST + n] = (_Float16)fmaxf(acc[r] + bn, 0.0f);
  }
}

// Projection: D(row-major) = relu(A0*B0 + A1*B1 + bias)   (concat -> split GEMM)
__device__ __forceinline__ void gemm_proj(const _Float16* __restrict__ A0,
                                          const _Float16* __restrict__ B0,
                                          const _Float16* __restrict__ A1,
                                          const _Float16* __restrict__ B1,
                                          const float* __restrict__ bias,
                                          _Float16* __restrict__ D, int wave, int lane) {
#pragma unroll
  for (int nt = 0; nt < 8; ++nt) {
    v8f acc = {0.f, 0.f, 0.f, 0.f, 0.f, 0.f, 0.f, 0.f};
#pragma unroll
    for (int kt = 0; kt < 4; ++kt) {
      v16h a = load_frag(A0, wave, kt, lane, LDST);
      v16h b = load_frag(B0, nt,  kt, lane, 128);
      acc = __builtin_amdgcn_wmma_f32_16x16x32_f16(false, a, false, b, (short)0, acc, false, false);
    }
#pragma unroll
    for (int kt = 0; kt < 4; ++kt) {
      v16h a = load_frag(A1, wave, kt, lane, LDST);
      v16h b = load_frag(B1, nt,  kt, lane, 128);
      acc = __builtin_amdgcn_wmma_f32_16x16x32_f16(false, a, false, b, (short)0, acc, false, false);
    }
    const int n  = nt * 16 + (lane & 15);
    const int m0 = wave * 16 + ((lane >> 4) & 1) * 8;
    const float bn = bias[n];
#pragma unroll
    for (int r = 0; r < 8; ++r)
      D[(m0 + r) * LDST + n] = (_Float16)fmaxf(acc[r] + bn, 0.0f);
  }
}

// ---------- Kernel 0: f32 -> f16 conversion + weight transpose (N-major) ----------
__global__ __launch_bounds__(256)
void gcnmix_prep(const float* __restrict__ atom_emb, const float* __restrict__ fp_emb,
                 const float* __restrict__ W_proj, const float* __restrict__ gcn_W,
                 _Float16* __restrict__ oAtom, _Float16* __restrict__ oFp,
                 _Float16* __restrict__ oW1, _Float16* __restrict__ oW2,
                 _Float16* __restrict__ oG0, _Float16* __restrict__ oG1) {
  const int t = blockIdx.x * blockDim.x + threadIdx.x;
  if (t < 100 * 128) oAtom[t] = (_Float16)atom_emb[t];
  if (t < 2048 * 128) oFp[t] = (_Float16)fp_emb[t];
  if (t < 128 * 128) {
    const int o = t >> 7, i = t & 127;            // transposed: [out][in]
    oW1[t] = (_Float16)W_proj[i * 128 + o];
    oW2[t] = (_Float16)W_proj[(128 + i) * 128 + o];
    oG0[t] = (_Float16)gcn_W[i * 128 + o];
    oG1[t] = (_Float16)gcn_W[16384 + i * 128 + o];
  }
}

// ---------- Kernel 1: per-(b,m) graph: gather + Anorm + proj + 2 GCN layers + pool ----------
__global__ __launch_bounds__(256)
void gcnmix_graph(const int* __restrict__ af, const int* __restrict__ fp,
                  const int* __restrict__ adj,
                  const _Float16* __restrict__ atomE, const _Float16* __restrict__ fpE,
                  const _Float16* __restrict__ W1T, const _Float16* __restrict__ W2T,
                  const _Float16* __restrict__ G0T, const _Float16* __restrict__ G1T,
                  const float* __restrict__ b_proj, const float* __restrict__ gcn_b,
                  float* __restrict__ mol) {
  extern __shared__ _Float16 smem[];
  _Float16* s0 = smem;               // atom gather -> layer0 output x2 -> (free)
  _Float16* s1 = smem + LDS_MAT;     // fp gather   -> xW intermediates (col-major)
  _Float16* s2 = smem + 2 * LDS_MAT; // x (proj out) -> final x
  _Float16* s3 = smem + 3 * LDS_MAT; // Anorm (persistent, f16)
  __shared__ float dinv[128];

  const int bm   = blockIdx.x;       // b*M + m
  const int tid  = threadIdx.x;
  const int wave = tid >> 5;
  const int lane = tid & 31;

  // --- gather atom/fp embedding rows (2 threads per node, 64 halves each) ---
  {
    const int node = tid >> 1, hf = tid & 1;
    const int ai = af[bm * NN + node];
    const int fi = fp[bm * NN + node];
    const _Float16* srcA = atomE + (size_t)ai * H + hf * 64;
    const _Float16* srcF = fpE   + (size_t)fi * H + hf * 64;
    _Float16* dA = s0 + node * LDST + hf * 64;
    _Float16* dF = s1 + node * LDST + hf * 64;
#pragma unroll
    for (int k = 0; k < 8; ++k) {
      *(v8h*)(dA + k * 8) = *(const v8h*)(srcA + k * 8);
      *(v8h*)(dF + k * 8) = *(const v8h*)(srcF + k * 8);
    }
  }
  // --- adj -> raw A entries, fully coalesced int4 (b128) streaming ---
  // A = (adj != 0) + 2*I ; 16 iterations x 256 lanes x int4 covers 128x128 ints.
  {
    const int* adjblk = adj + (size_t)bm * NN * NN;
#pragma unroll
    for (int it = 0; it < 16; ++it) {
      const int c  = tid + it * 256;         // int4 chunk id, 0..4095
      const int4 q = ((const int4*)adjblk)[c];
      const int i  = c >> 5;                 // row
      const int j0 = (c & 31) * 4;           // col base
      v4h o;
      o[0] = (_Float16)(float)((q.x != 0 ? 1 : 0) + (j0 + 0 == i ? 2 : 0));
      o[1] = (_Float16)(float)((q.y != 0 ? 1 : 0) + (j0 + 1 == i ? 2 : 0));
      o[2] = (_Float16)(float)((q.z != 0 ? 1 : 0) + (j0 + 2 == i ? 2 : 0));
      o[3] = (_Float16)(float)((q.w != 0 ? 1 : 0) + (j0 + 3 == i ? 2 : 0));
      *(v4h*)(s3 + i * LDST + j0) = o;       // 8B ds_store, aligned
    }
  }
  __syncthreads();
  // --- degrees from LDS (vector row sums), dinv = rsqrt(deg) ---
  if (tid < 128) {
    float s = 0.f;
#pragma unroll
    for (int k = 0; k < 16; ++k) {
      v8h v = *(const v8h*)(s3 + tid * LDST + k * 8);
#pragma unroll
      for (int e = 0; e < 8; ++e) s += (float)v[e];
    }
    dinv[tid] = rsqrtf(s);
  }
  __syncthreads();
  // --- symmetric normalization ---
  for (int e = tid; e < NN * NN; e += 256) {
    const int i = e >> 7, j = e & 127;
    s3[i * LDST + j] = (_Float16)((float)s3[i * LDST + j] * dinv[i] * dinv[j]);
  }
  __syncthreads();

  // --- projection: x = relu([atom|fp] @ W_proj + b) ---
  gemm_proj(s0, W1T, s1, W2T, b_proj, s2, wave, lane);
  __syncthreads();
  // --- layer 0: xW = x @ W0 (col-major), x2 = relu(Anorm @ xW + b0) ---
  gemm_colmajor(s2, LDST, G0T, 128, s1, wave, lane);
  __syncthreads();
  gemm_bias_relu_row(s3, LDST, s1, LDST, gcn_b, s0, wave, lane);
  __syncthreads();
  // --- layer 1 ---
  gemm_colmajor(s0, LDST, G1T, 128, s1, wave, lane);
  __syncthreads();
  gemm_bias_relu_row(s3, LDST, s1, LDST, gcn_b + 128, s2, wave, lane);
  __syncthreads();

  // --- mean pool over nodes ---
  if (tid < 128) {
    float s = 0.f;
    for (int node = 0; node < NN; ++node) s += (float)s2[node * LDST + tid];
    mol[(size_t)bm * H + tid] = s * (1.0f / 128.0f);
  }
}

__device__ __forceinline__ float nan2num(float v, float pos, float neg) {
  if (v != v) return 0.f;
  if (v == __builtin_inff()) return pos;
  if (v == -__builtin_inff()) return neg;
  return v;
}

// ---------- Kernel 2: per-b mixing + LayerNorm + MLP head ----------
__global__ __launch_bounds__(128)
void gcnmix_head(const float* __restrict__ mol, const float* __restrict__ phys,
                 const float* __restrict__ ratios,
                 const float* __restrict__ ln_g, const float* __restrict__ ln_b,
                 const float* __restrict__ W_h1, const float* __restrict__ b_h1,
                 const float* __restrict__ W_h2, const float* __restrict__ b_h2,
                 float* __restrict__ out) {
  const int b = blockIdx.x, t = threadIdx.x;
  __shared__ float z[144];
  __shared__ float red[128];

  float w[4];
  {
    float rs = 1e-8f;
#pragma unroll
    for (int m = 0; m < 4; ++m) { w[m] = ratios[b * 4 + m]; rs += w[m]; }
    const float inv = 1.0f / rs;
#pragma unroll
    for (int m = 0; m < 4; ++m) w[m] *= inv;
  }
  {
    float s = 0.f;
#pragma unroll
    for (int m = 0; m < 4; ++m) s += w[m] * mol[((size_t)b * 4 + m) * H + t];
    z[t] = s;
  }
  if (t < 16) {
    float s = 0.f;
#pragma unroll
    for (int m = 0; m < 4; ++m)
      s += w[m] * nan2num(phys[((size_t)b * 4 + m) * 16 + t], 1000.f, -1000.f);
    z[128 + t] = s;
  }
  __syncthreads();

  float p1 = z[t], p2 = z[t] * z[t];
  if (t < 16) { const float e = z[128 + t]; p1 += e; p2 += e * e; }
  red[t] = p1; __syncthreads();
  for (int s = 64; s > 0; s >>= 1) { if (t < s) red[t] += red[t + s]; __syncthreads(); }
  const float mu = red[0] * (1.0f / 144.0f);
  __syncthreads();
  red[t] = p2; __syncthreads();
  for (int s = 64; s > 0; s >>= 1) { if (t < s) red[t] += red[t + s]; __syncthreads(); }
  const float var  = red[0] * (1.0f / 144.0f) - mu * mu;
  const float rstd = rsqrtf(var + 1e-5f);
  __syncthreads();

  const float zt = (z[t] - mu) * rstd * ln_g[t] + ln_b[t];
  float zh = 0.f;
  if (t < 16) zh = (z[128 + t] - mu) * rstd * ln_g[128 + t] + ln_b[128 + t];
  __syncthreads();
  z[t] = zt;
  if (t < 16) z[128 + t] = zh;
  __syncthreads();

  float acc = b_h1[t];
  for (int k = 0; k < 144; ++k) acc += z[k] * W_h1[k * 128 + t];
  red[t] = fmaxf(acc, 0.f) * W_h2[t];
  __syncthreads();
  for (int s = 64; s > 0; s >>= 1) { if (t < s) red[t] += red[t + s]; __syncthreads(); }
  if (t == 0)
    out[b] = nan2num(red[0] + b_h2[0], 3.402823466e38f, -3.402823466e38f);
}

extern "C" void kernel_launch(void* const* d_in, const int* in_sizes, int n_in,
                              void* d_out, int out_size, void* d_ws, size_t ws_size,
                              hipStream_t stream) {
  const int*   af       = (const int*)d_in[0];
  const int*   fpi      = (const int*)d_in[1];
  const int*   adj      = (const int*)d_in[2];
  const float* phys     = (const float*)d_in[3];
  const float* ratios   = (const float*)d_in[4];
  const float* atom_emb = (const float*)d_in[5];
  const float* fp_emb   = (const float*)d_in[6];
  const float* W_proj   = (const float*)d_in[7];
  const float* b_proj   = (const float*)d_in[8];
  const float* gcn_W    = (const float*)d_in[9];
  const float* gcn_b    = (const float*)d_in[10];
  const float* ln_g     = (const float*)d_in[11];
  const float* ln_b     = (const float*)d_in[12];
  const float* W_h1     = (const float*)d_in[13];
  const float* b_h1     = (const float*)d_in[14];
  const float* W_h2     = (const float*)d_in[15];
  const float* b_h2     = (const float*)d_in[16];

  _Float16* hws = (_Float16*)d_ws;
  const size_t oAtom = 0;
  const size_t oFp   = oAtom + 100 * 128;
  const size_t oW1   = oFp + 2048 * 128;
  const size_t oW2   = oW1 + 16384;
  const size_t oG0   = oW2 + 16384;
  const size_t oG1   = oG0 + 16384;
  const size_t endH  = oG1 + 16384;
  float* mol = (float*)((char*)d_ws + ((endH * 2 + 255) / 256) * 256);

  gcnmix_prep<<<1024, 256, 0, stream>>>(atom_emb, fp_emb, W_proj, gcn_W,
                                        hws + oAtom, hws + oFp, hws + oW1,
                                        hws + oW2, hws + oG0, hws + oG1);
  gcnmix_graph<<<1024, 256, LDS_BYTES, stream>>>(af, fpi, adj,
                                                 hws + oAtom, hws + oFp,
                                                 hws + oW1, hws + oW2,
                                                 hws + oG0, hws + oG1,
                                                 b_proj, gcn_b, mol);
  gcnmix_head<<<256, 128, 0, stream>>>(mol, phys, ratios, ln_g, ln_b,
                                       W_h1, b_h1, W_h2, b_h2, (float*)d_out);
}
